// GatedGCNModel_73443940762179
// MI455X (gfx1250) — matrix operand order repaired and checked by reading, not compiled, not run on hardware
//
#include <hip/hip_runtime.h>
#include <hip/hip_bf16.h>

typedef __attribute__((ext_vector_type(16))) _Float16 v16h;
typedef __attribute__((ext_vector_type(8)))  float    v8f;

#define N_NODES 50000
#define N_EDGES 400000
#define EPS 1e-5f

__device__ __forceinline__ v8f wmma16(v16h a, v16h b, v8f c) {
    // D = A(16x32 f16) * B(32x16 f16) + C(16x16 f32)
    return __builtin_amdgcn_wmma_f32_16x16x32_f16(
        /*neg_a=*/false, a, /*neg_b=*/false, b,
        /*c_mod=*/(short)0, c, /*reuse_a=*/false, /*reuse_b=*/false);
}

// Load A fragment (16x32, f16) from a row-major f32 matrix, converting on the fly.
// ISA layout: lanes 0-15 -> M=0..15 (half=0), lanes 16-31 -> M=0..15 (half=1).
// VGPR v holds K-pair kb,kb+1 with kb = (v<4 ? 2v : 8+2v) + 8*half.
__device__ __forceinline__ v16h load_a_f32(const float* __restrict__ A, int lda,
                                           int row0, int k0, int lane) {
    int m = lane & 15;
    int half = lane >> 4;
    const float* rp = A + (size_t)(row0 + m) * lda + k0;
    v16h a;
#pragma unroll
    for (int v = 0; v < 8; ++v) {
        int kb = (v < 4 ? v * 2 : 8 + v * 2) + half * 8;
        float2 p2 = *(const float2*)(rp + kb);
        a[2 * v]     = (_Float16)p2.x;
        a[2 * v + 1] = (_Float16)p2.y;
    }
    return a;
}

// ---------------------------------------------------------------------------
// K=64 WMMA GEMM: Y[M, NT*16] = Af32[M,64] @ Wf16[64, NT*16] + bias.
// NT is a template constant so C-stores use immediate offsets. Each wave
// computes TWO 16-row M tiles, reusing each loaded B fragment for 2 WMMAs.
// ---------------------------------------------------------------------------
template <int NT>
__global__ void __launch_bounds__(256) gemm_k64(const float* __restrict__ A,
                                                const _Float16* __restrict__ W,
                                                const float* __restrict__ bias,
                                                float* __restrict__ Y,
                                                int mTiles) {
    constexpr int LDW = NT * 16;
    int wv = (int)((blockIdx.x * blockDim.x + threadIdx.x) >> 5);
    int t0 = wv * 2;
    if (t0 >= mTiles) return;
    bool has2 = (t0 + 1) < mTiles;
    int lane = threadIdx.x & 31;
    int row0 = t0 * 16;

    v16h a00 = load_a_f32(A, 64, row0, 0, lane);
    v16h a01 = load_a_f32(A, 64, row0, 32, lane);
    v16h a10, a11;
    if (has2) {
        a10 = load_a_f32(A, 64, row0 + 16, 0, lane);
        a11 = load_a_f32(A, 64, row0 + 16, 32, lane);
    }

    int n_lo = lane & 15;
    int mbase = row0 + 8 * (lane >> 4);

#pragma unroll
    for (int nt = 0; nt < NT; ++nt) {
        int n0 = nt * 16;
        // B layout: lane -> N = n0 + lane%16; element s -> K = (s<8 ? s : 8+s) + 8*(lane/16).
        v16h b0, b1;
#pragma unroll
        for (int s = 0; s < 16; ++s) {
            int kk = (s < 8 ? s : 8 + s) + 8 * (lane >> 4);
            b0[s] = W[(size_t)kk * LDW + n0 + (lane & 15)];
            b1[s] = W[(size_t)(32 + kk) * LDW + n0 + (lane & 15)];
        }

        v8f c0 = {0.f, 0.f, 0.f, 0.f, 0.f, 0.f, 0.f, 0.f};
        c0 = wmma16(a00, b0, c0);
        c0 = wmma16(a01, b1, c0);

        float bn = bias[n0 + n_lo];
        float* yp = Y + (size_t)mbase * LDW + n0 + n_lo;
#pragma unroll
        for (int r = 0; r < 8; ++r) yp[(size_t)r * LDW] = c0[r] + bn;

        if (has2) {
            v8f c1 = {0.f, 0.f, 0.f, 0.f, 0.f, 0.f, 0.f, 0.f};
            c1 = wmma16(a10, b0, c1);
            c1 = wmma16(a11, b1, c1);
            float* yq = yp + (size_t)16 * LDW;
#pragma unroll
            for (int r = 0; r < 8; ++r) yq[(size_t)r * LDW] = c1[r] + bn;
        }
    }
}

// ---------------------------------------------------------------------------
// Predictor GEMM: t1[E,64] = cat(x[src], x[dst], e)[E,192] @ pW1f16[192,64] + pb1
// Gather is done in the A-fragment loader; k-tile boundaries (32) align with
// the 64-wide concat segments so each k-tile reads from a single source.
// ---------------------------------------------------------------------------
__global__ void __launch_bounds__(256) pred_gemm(const float* __restrict__ x,
                                                 const float* __restrict__ e,
                                                 const int* __restrict__ src,
                                                 const int* __restrict__ dst,
                                                 const _Float16* __restrict__ W,
                                                 const float* __restrict__ bias,
                                                 float* __restrict__ Y, int mTiles) {
    int wv = (int)((blockIdx.x * blockDim.x + threadIdx.x) >> 5);
    if (wv >= mTiles) return;
    int lane = threadIdx.x & 31;
    int row0 = wv * 16;
    int m = lane & 15;
    int half = lane >> 4;
    int row = row0 + m;
    int s = src[row], d = dst[row];
    __builtin_prefetch(x + (size_t)s * 64);
    __builtin_prefetch(x + (size_t)d * 64);

    v16h af[6];
#pragma unroll
    for (int kt = 0; kt < 6; ++kt) {
        int k0 = kt * 32;
        const float* base = (k0 < 64)  ? (x + (size_t)s * 64 + k0)
                          : (k0 < 128) ? (x + (size_t)d * 64 + (k0 - 64))
                                       : (e + (size_t)row * 64 + (k0 - 128));
        v16h a;
#pragma unroll
        for (int v = 0; v < 8; ++v) {
            int kb = (v < 4 ? v * 2 : 8 + v * 2) + half * 8;
            float2 p2 = *(const float2*)(base + kb);
            a[2 * v]     = (_Float16)p2.x;
            a[2 * v + 1] = (_Float16)p2.y;
        }
        af[kt] = a;
    }

    int n_lo = lane & 15;
    int mbase = row0 + 8 * (lane >> 4);
#pragma unroll
    for (int nt = 0; nt < 4; ++nt) {
        int n0 = nt * 16;
        v8f c = {0.f, 0.f, 0.f, 0.f, 0.f, 0.f, 0.f, 0.f};
#pragma unroll
        for (int kt = 0; kt < 6; ++kt) {
            v16h b;
#pragma unroll
            for (int s = 0; s < 16; ++s)
                b[s] = W[(size_t)(kt * 32 + (s < 8 ? s : 8 + s) + 8 * half) * 64 + n0 + (lane & 15)];
            c = wmma16(af[kt], b, c);
        }
        float bn = bias[n0 + n_lo];
        float* yp = Y + (size_t)mbase * 64 + n0 + n_lo;
#pragma unroll
        for (int r = 0; r < 8; ++r) yp[(size_t)r * 64] = c[r] + bn;
    }
}

// ---------------------------------------------------------------------------
// 2-layer MLP encoder (in [M,2] -> out [M,64]). One wave per row, 2 feats/lane.
// ---------------------------------------------------------------------------
__global__ void __launch_bounds__(256) encode_mlp(const float* __restrict__ in,
                                                  const float* __restrict__ W1,
                                                  const float* __restrict__ b1,
                                                  const float* __restrict__ W2,
                                                  const float* __restrict__ b2,
                                                  float* __restrict__ out, int M) {
    int wv = (int)((blockIdx.x * blockDim.x + threadIdx.x) >> 5);
    if (wv >= M) return;
    int lane = threadIdx.x & 31;
    float i0 = in[(size_t)wv * 2];
    float i1 = in[(size_t)wv * 2 + 1];
    float h[16];
#pragma unroll
    for (int j = 0; j < 16; ++j)
        h[j] = fmaxf(fmaf(i1, W1[16 + j], fmaf(i0, W1[j], b1[j])), 0.f);
    int f0 = lane * 2;
    float o0 = b2[f0], o1 = b2[f0 + 1];
#pragma unroll
    for (int j = 0; j < 16; ++j) {
        o0 = fmaf(h[j], W2[j * 64 + f0], o0);
        o1 = fmaf(h[j], W2[j * 64 + f0 + 1], o1);
    }
    out[(size_t)wv * 64 + f0] = o0;
    out[(size_t)wv * 64 + f0 + 1] = o1;
}

// ---------------------------------------------------------------------------
// Weight prep: f32 -> f16 packs. W5[l][64][320] = [A1|A2|A3|B1|B2], b5[l][320],
// B3h[l][64][64], pW1h[192][64].
// ---------------------------------------------------------------------------
__global__ void prep_weights(const float* __restrict__ A1, const float* __restrict__ A2,
                             const float* __restrict__ A3, const float* __restrict__ B1,
                             const float* __restrict__ B2, const float* __restrict__ B3,
                             const float* __restrict__ bA1, const float* __restrict__ bA2,
                             const float* __restrict__ bA3, const float* __restrict__ bB1,
                             const float* __restrict__ bB2, const float* __restrict__ pW1,
                             _Float16* __restrict__ W5, _Float16* __restrict__ B3h,
                             _Float16* __restrict__ pW1h, float* __restrict__ b5) {
    int idx = blockIdx.x * blockDim.x + threadIdx.x;
    const int T1 = 4 * 64 * 320, T2 = 4 * 64 * 64, T3 = 192 * 64, T4 = 4 * 320;
    if (idx < T1) {
        int l = idx / (64 * 320);
        int r = idx % (64 * 320);
        int k = r / 320, j = r % 320;
        int blk = j >> 6, jj = j & 63;
        const float* M = (blk == 0) ? A1 : (blk == 1) ? A2 : (blk == 2) ? A3
                        : (blk == 3) ? B1 : B2;
        W5[idx] = (_Float16)M[(size_t)l * 4096 + k * 64 + jj];
    } else if ((idx -= T1) < T2) {
        B3h[idx] = (_Float16)B3[idx];
    } else if ((idx -= T2) < T3) {
        pW1h[idx] = (_Float16)pW1[idx];
    } else if ((idx -= T3) < T4) {
        int l = idx / 320, j = idx % 320;
        int blk = j >> 6, jj = j & 63;
        const float* bv = (blk == 0) ? bA1 : (blk == 1) ? bA2 : (blk == 2) ? bA3
                         : (blk == 3) ? bB1 : bB2;
        b5[idx] = bv[l * 64 + jj];
    }
}

// ---------------------------------------------------------------------------
// Edge update: e += ReLU(LN(B1 x_s + B2 x_d + B3 e)); sigma = sigmoid(e);
// scatter-accumulate m_f/d_f/m_b/d_b with f32 global atomics.
// One wave per edge; 2 feats/lane; LN reductions via wave32 shfl_xor.
// ---------------------------------------------------------------------------
__global__ void __launch_bounds__(256) edge_update(const float* __restrict__ xcat,
                                                   const float* __restrict__ eB3,
                                                   float* __restrict__ e,
                                                   const int* __restrict__ src,
                                                   const int* __restrict__ dst,
                                                   const float* __restrict__ ge,
                                                   const float* __restrict__ be,
                                                   float* __restrict__ agg,
                                                   int nEdges, int nNodes) {
    int wv = (int)((blockIdx.x * blockDim.x + threadIdx.x) >> 5);
    if (wv >= nEdges) return;
    int lane = threadIdx.x & 31;
    int f0 = lane * 2;
    int s = src[wv], d = dst[wv];
    const float* xs = xcat + (size_t)s * 320;
    const float* xd = xcat + (size_t)d * 320;
    __builtin_prefetch(xs + 64);
    __builtin_prefetch(xd + 128);
    const float* eb = eB3 + (size_t)wv * 64;

    float2 vs = *(const float2*)(xs + 192 + f0);  // (x @ B1 + bB1)[src]
    float2 vd = *(const float2*)(xd + 256 + f0);  // (x @ B2 + bB2)[dst]
    float2 vb = *(const float2*)(eb + f0);        // e @ B3 + bB3
    float z0 = vs.x + vd.x + vb.x;
    float z1 = vs.y + vd.y + vb.y;

    float s1 = z0 + z1, s2 = z0 * z0 + z1 * z1;
#pragma unroll
    for (int msk = 16; msk >= 1; msk >>= 1) {
        s1 += __shfl_xor(s1, msk, 32);
        s2 += __shfl_xor(s2, msk, 32);
    }
    float mu = s1 * (1.f / 64.f);
    float var = s2 * (1.f / 64.f) - mu * mu;
    float inv = rsqrtf(var + EPS);

    float l0 = (z0 - mu) * inv * ge[f0] + be[f0];
    float l1 = (z1 - mu) * inv * ge[f0 + 1] + be[f0 + 1];

    float* ep = e + (size_t)wv * 64;
    float2 eo = *(const float2*)(ep + f0);
    float e0 = eo.x + fmaxf(l0, 0.f);
    float e1 = eo.y + fmaxf(l1, 0.f);
    *(float2*)(ep + f0) = make_float2(e0, e1);

    float g0 = 1.f / (1.f + __expf(-e0));
    float g1 = 1.f / (1.f + __expf(-e1));

    float2 a2s = *(const float2*)(xs + 64 + f0);   // (x @ A2 + bA2)[src]
    float2 a3d = *(const float2*)(xd + 128 + f0);  // (x @ A3 + bA3)[dst]

    size_t NF = (size_t)nNodes * 64;
    float* m_f = agg;
    float* d_f = agg + NF;
    float* m_b = agg + 2 * NF;
    float* d_b = agg + 3 * NF;
    size_t od = (size_t)d * 64 + f0;
    size_t os = (size_t)s * 64 + f0;
    unsafeAtomicAdd(&m_f[od],     g0 * a2s.x);
    unsafeAtomicAdd(&m_f[od + 1], g1 * a2s.y);
    unsafeAtomicAdd(&d_f[od],     g0);
    unsafeAtomicAdd(&d_f[od + 1], g1);
    unsafeAtomicAdd(&m_b[os],     g0 * a3d.x);
    unsafeAtomicAdd(&m_b[os + 1], g1 * a3d.y);
    unsafeAtomicAdd(&d_b[os],     g0);
    unsafeAtomicAdd(&d_b[os + 1], g1);
}

// ---------------------------------------------------------------------------
// Node update: x += ReLU(LN(A1 x + m_f/(d_f+eps) + m_b/(d_b+eps)))
// ---------------------------------------------------------------------------
__global__ void __launch_bounds__(256) node_update(const float* __restrict__ xcat,
                                                   const float* __restrict__ agg,
                                                   const float* __restrict__ gh,
                                                   const float* __restrict__ bh,
                                                   float* __restrict__ x, int nNodes) {
    int wv = (int)((blockIdx.x * blockDim.x + threadIdx.x) >> 5);
    if (wv >= nNodes) return;
    int lane = threadIdx.x & 31;
    int f0 = lane * 2;
    size_t NF = (size_t)nNodes * 64;
    const float* m_f = agg;
    const float* d_f = agg + NF;
    const float* m_b = agg + 2 * NF;
    const float* d_b = agg + 3 * NF;
    size_t o = (size_t)wv * 64 + f0;
    float2 a1 = *(const float2*)(xcat + (size_t)wv * 320 + f0);  // x @ A1 + bA1
    float h0 = a1.x + m_f[o]     / (d_f[o]     + EPS) + m_b[o]     / (d_b[o]     + EPS);
    float h1 = a1.y + m_f[o + 1] / (d_f[o + 1] + EPS) + m_b[o + 1] / (d_b[o + 1] + EPS);

    float s1 = h0 + h1, s2 = h0 * h0 + h1 * h1;
#pragma unroll
    for (int msk = 16; msk >= 1; msk >>= 1) {
        s1 += __shfl_xor(s1, msk, 32);
        s2 += __shfl_xor(s2, msk, 32);
    }
    float mu = s1 * (1.f / 64.f);
    float var = s2 * (1.f / 64.f) - mu * mu;
    float inv = rsqrtf(var + EPS);
    float l0 = (h0 - mu) * inv * gh[f0] + bh[f0];
    float l1 = (h1 - mu) * inv * gh[f0 + 1] + bh[f0 + 1];
    float* xp = x + (size_t)wv * 64;
    float2 xo = *(const float2*)(xp + f0);
    *(float2*)(xp + f0) = make_float2(xo.x + fmaxf(l0, 0.f), xo.y + fmaxf(l1, 0.f));
}

// ---------------------------------------------------------------------------
// Final score: out[i] = relu(t1[i,:]) . pW2 + pb2. One wave per edge.
// ---------------------------------------------------------------------------
__global__ void __launch_bounds__(256) score_out(const float* __restrict__ t1,
                                                 const float* __restrict__ pW2,
                                                 const float* __restrict__ pb2,
                                                 float* __restrict__ out, int nEdges) {
    int wv = (int)((blockIdx.x * blockDim.x + threadIdx.x) >> 5);
    if (wv >= nEdges) return;
    int lane = threadIdx.x & 31;
    int f0 = lane * 2;
    const float* tp = t1 + (size_t)wv * 64;
    float2 t = *(const float2*)(tp + f0);
    float v = fmaxf(t.x, 0.f) * pW2[f0] + fmaxf(t.y, 0.f) * pW2[f0 + 1];
#pragma unroll
    for (int msk = 16; msk >= 1; msk >>= 1) v += __shfl_xor(v, msk, 32);
    if (lane == 0) out[wv] = v + pb2[0];
}

extern "C" void kernel_launch(void* const* d_in, const int* in_sizes, int n_in,
                              void* d_out, int out_size, void* d_ws, size_t ws_size,
                              hipStream_t stream) {
    const int N = N_NODES, E = N_EDGES;
    const float* x2   = (const float*)d_in[0];
    const float* e2   = (const float*)d_in[1];
    const int*   src  = (const int*)d_in[2];
    const int*   dst  = (const int*)d_in[3];
    const float* neW1 = (const float*)d_in[4];
    const float* neb1 = (const float*)d_in[5];
    const float* neW2 = (const float*)d_in[6];
    const float* neb2 = (const float*)d_in[7];
    const float* eeW1 = (const float*)d_in[8];
    const float* eeb1 = (const float*)d_in[9];
    const float* eeW2 = (const float*)d_in[10];
    const float* eeb2 = (const float*)d_in[11];
    const float* A1   = (const float*)d_in[12];
    const float* bA1  = (const float*)d_in[13];
    const float* A2   = (const float*)d_in[14];
    const float* bA2  = (const float*)d_in[15];
    const float* A3   = (const float*)d_in[16];
    const float* bA3  = (const float*)d_in[17];
    const float* B1   = (const float*)d_in[18];
    const float* bB1  = (const float*)d_in[19];
    const float* B2   = (const float*)d_in[20];
    const float* bB2  = (const float*)d_in[21];
    const float* B3   = (const float*)d_in[22];
    const float* bB3  = (const float*)d_in[23];
    const float* ge   = (const float*)d_in[24];
    const float* be   = (const float*)d_in[25];
    const float* gh   = (const float*)d_in[26];
    const float* bh   = (const float*)d_in[27];
    const float* pW1  = (const float*)d_in[28];
    const float* pb1  = (const float*)d_in[29];
    const float* pW2  = (const float*)d_in[30];
    const float* pb2  = (const float*)d_in[31];
    float* out = (float*)d_out;

    // ---- workspace carve-up (256B aligned) ----
    char* wp = (char*)d_ws;
    auto carve = [&](size_t bytes) -> void* {
        void* r = (void*)wp;
        wp += (bytes + 255) & ~(size_t)255;
        return r;
    };
    float*    x_cur = (float*)carve((size_t)N * 64 * 4);        // node features
    float*    e_cur = (float*)carve((size_t)E * 64 * 4);        // edge features
    float*    xcat  = (float*)carve((size_t)N * 320 * 4);       // [xA1|xA2|xA3|xB1|xB2]
    float*    t1    = (float*)carve((size_t)E * 64 * 4);        // eB3 / predictor hidden
    float*    agg   = (float*)carve((size_t)4 * N * 64 * 4);    // m_f,d_f,m_b,d_b
    _Float16* W5    = (_Float16*)carve((size_t)4 * 64 * 320 * 2);
    _Float16* B3h   = (_Float16*)carve((size_t)4 * 64 * 64 * 2);
    _Float16* pW1h  = (_Float16*)carve((size_t)192 * 64 * 2);
    float*    b5    = (float*)carve((size_t)4 * 320 * 4);

    // ---- weight prep + encoders ----
    {
        int total = 4 * 64 * 320 + 4 * 64 * 64 + 192 * 64 + 4 * 320;
        prep_weights<<<(total + 255) / 256, 256, 0, stream>>>(
            A1, A2, A3, B1, B2, B3, bA1, bA2, bA3, bB1, bB2, pW1, W5, B3h, pW1h, b5);
    }
    encode_mlp<<<(N + 7) / 8, 256, 0, stream>>>(x2, neW1, neb1, neW2, neb2, x_cur, N);
    encode_mlp<<<(E + 7) / 8, 256, 0, stream>>>(e2, eeW1, eeb1, eeW2, eeb2, e_cur, E);

    const int mTilesN = N / 16;   // 3125
    const int mTilesE = E / 16;   // 25000
    // each gemm wave handles 2 M-tiles; 8 waves per block
    const int wavesN = (mTilesN + 1) / 2;
    const int wavesE = (mTilesE + 1) / 2;

    for (int l = 0; l < 4; ++l) {
        // fused node transform: xcat = x @ [A1|A2|A3|B1|B2] + biases  (WMMA)
        gemm_k64<20><<<(wavesN + 7) / 8, 256, 0, stream>>>(
            x_cur, W5 + (size_t)l * 64 * 320, b5 + l * 320, xcat, mTilesN);
        // edge transform: t1 = e @ B3 + bB3  (WMMA)
        gemm_k64<4><<<(wavesE + 7) / 8, 256, 0, stream>>>(
            e_cur, B3h + (size_t)l * 64 * 64, bB3 + l * 64, t1, mTilesE);
        hipMemsetAsync(agg, 0, (size_t)4 * N * 64 * sizeof(float), stream);
        edge_update<<<E / 8, 256, 0, stream>>>(
            xcat, t1, e_cur, src, dst, ge + l * 64, be + l * 64, agg, E, N);
        node_update<<<N / 8, 256, 0, stream>>>(
            xcat, agg, gh + l * 64, bh + l * 64, x_cur, N);
    }

    // predictor
    pred_gemm<<<(mTilesE + 7) / 8, 256, 0, stream>>>(
        x_cur, e_cur, src, dst, pW1h, pb1, t1, mTilesE);
    score_out<<<E / 8, 256, 0, stream>>>(t1, pW2, pb2, out, E);
}
